// CustomEmbed_70076686402000
// MI455X (gfx1250) — matrix operand order repaired and checked by reading, not compiled, not run on hardware
//
#include <hip/hip_runtime.h>
#include <hip/hip_bf16.h>
#include <math.h>

typedef _Float16 f16;
typedef __attribute__((ext_vector_type(16))) _Float16 v16h;
typedef __attribute__((ext_vector_type(8)))  float    v8f;
typedef int v4i __attribute__((vector_size(16)));

#define EPSBN 1e-5f

// padded LDS row strides (in f16) to dodge bank conflicts on 16B lane loads
#define ASTR 40
#define BSTR 40

#if __has_builtin(__builtin_amdgcn_global_load_async_to_lds_b128)
#define HAVE_ASYNC_LDS 1
#else
#define HAVE_ASYNC_LDS 0
#endif

__device__ __forceinline__ void async_b128_to_lds(const f16* gsrc, f16* ldst) {
#if HAVE_ASYNC_LDS
    __builtin_amdgcn_global_load_async_to_lds_b128((v4i*)gsrc, (v4i*)ldst, 0, 0);
#else
    *(uint4*)ldst = *(const uint4*)gsrc;
#endif
}

__device__ __forceinline__ void wait_async_done() {
#if __has_builtin(__builtin_amdgcn_s_wait_asynccnt)
    __builtin_amdgcn_s_wait_asynccnt(0);
#else
    asm volatile("s_wait_asynccnt 0x0" ::: "memory");
#endif
}

__device__ __forceinline__ float gelu_exact(float x) {
    return 0.5f * x * (1.0f + erff(x * 0.70710678118654752f));
}

union Frag16 { v16h v; uint4 q[2]; };
union Pack8  { f16 h[8]; uint4 q; };

// ---------------------------------------------------------------------------
// Fold BN scale into conv weights (keep original (G, OC, K) layout, f16),
// emit per-(group, channel) bias.  K = IC*KH*KW.
// ---------------------------------------------------------------------------
__global__ void fold_weights(const float* __restrict__ w,
                             const float* __restrict__ g,
                             const float* __restrict__ b,
                             const float* __restrict__ m,
                             const float* __restrict__ v,
                             f16* __restrict__ wt, float* __restrict__ bias,
                             int G, int OC, int K) {
    size_t idx = (size_t)blockIdx.x * 256 + threadIdx.x;
    size_t total = (size_t)G * OC * K;
    if (idx >= total) return;
    int k = (int)(idx % K);
    size_t t = idx / K;
    int oc = (int)(t % OC);
    int grp = (int)(t / OC);
    float inv = g[(size_t)grp * OC + oc] * rsqrtf(v[(size_t)grp * OC + oc] + EPSBN);
    wt[idx] = (f16)(w[idx] * inv);
    if (k == 0)
        bias[(size_t)grp * OC + oc] =
            b[(size_t)grp * OC + oc] - m[(size_t)grp * OC + oc] * inv;
}

// ---------------------------------------------------------------------------
// conv1: 4x4 stride 4, 3->192, input (64,3,224,224) f32, out (64,192,56,56)
// f16 with BN+GELU applied.  K=48 too small for WMMA; direct VALU.
// wt layout [192][48] (k = ic*16 + dy*4 + dx), pre-scaled by BN.
// ---------------------------------------------------------------------------
__global__ void conv1_kernel(const float* __restrict__ x,
                             const f16* __restrict__ wt,
                             const float* __restrict__ bias,
                             f16* __restrict__ out) {
    size_t idx = (size_t)blockIdx.x * 256 + threadIdx.x;
    if (idx >= (size_t)64 * 192 * 56 * 56) return;
    int ox = (int)(idx % 56);
    int t  = (int)(idx / 56);
    int oy = t % 56; t /= 56;
    int oc = t % 192;
    int bb = t / 192;
    float s = 0.f;
    const float* xb = x + (size_t)bb * 3 * 224 * 224;
    const f16* wo = wt + oc * 48;
    #pragma unroll
    for (int ic = 0; ic < 3; ++ic) {
        const float* xc = xb + (size_t)ic * 224 * 224 + (size_t)(4 * oy) * 224 + 4 * ox;
        #pragma unroll
        for (int dy = 0; dy < 4; ++dy) {
            #pragma unroll
            for (int dx = 0; dx < 4; ++dx) {
                s += xc[dy * 224 + dx] * (float)wo[ic * 16 + dy * 4 + dx];
            }
        }
    }
    out[idx] = (f16)gelu_exact(s + bias[oc]);
}

// ---------------------------------------------------------------------------
// Region conv: per-tile 3x3 SAME conv 192->192 + BN + GELU + residual.
// Implicit GEMM, M = th*tw (49, padded to 64), N = C (192), K = C*9 (1728).
// One workgroup per (tile g, batch b). 8 waves; wave w: M-tile w>>1,
// 6 N-tiles from (w&1)*6.  LDS A: m-major/k-contig; LDS B: n-major/k-contig
// so every WMMA fragment is two ds_load_b128.  B tile staged with
// GLOBAL_LOAD_ASYNC_TO_LDS_B128 (ASYNCcnt) when the builtin is available.
// ---------------------------------------------------------------------------
__global__ __launch_bounds__(256)
void region_conv(const f16* __restrict__ hin, const f16* __restrict__ wt,
                 const float* __restrict__ bias, f16* __restrict__ hout,
                 int C, int H, int W, int gh, int gw) {
    const int th = H / gh, tw = W / gw;      // 7,7
    const int MT = th * tw;                  // 49
    const int K  = C * 9;                    // 1728
    const int g  = blockIdx.x;
    const int b  = blockIdx.y;
    const int ty0 = (g / gw) * th, tx0 = (g % gw) * tw;
    const int tid  = threadIdx.x;
    const int lane = tid & 31;
    const int wid  = tid >> 5;
    const int lm   = lane & 15;
    const bool hi  = (lane & 16) != 0;

    __shared__ __align__(16) f16 As[64 * ASTR];
    __shared__ __align__(16) f16 Bs[192 * BSTR];
    __shared__ int s_ic[32], s_dy[32], s_dx[32];

    const int mt  = wid >> 1;
    const int ntb = (wid & 1) * 6;

    v8f acc[6];
    #pragma unroll
    for (int t = 0; t < 6; ++t) {
        v8f z = {0.f, 0.f, 0.f, 0.f, 0.f, 0.f, 0.f, 0.f};
        acc[t] = z;
    }

    const f16* wtg = wt + (size_t)g * K * C;   // (OC=C rows) x (K cols)
    const size_t hbase = (size_t)b * C * H * W;

    for (int k0 = 0; k0 < K; k0 += 32) {
        __syncthreads();                     // previous compute done
        if (tid < 32) {
            int k = k0 + tid;
            int ic = k / 9;
            int r  = k - ic * 9;
            int dy = r / 3;
            s_ic[tid] = ic; s_dy[tid] = dy; s_dx[tid] = r - dy * 3;
        }
        // ---- stage B: async DMA 192 rows x 32 k-contiguous f16 to LDS ----
        {
            #pragma unroll
            for (int i = 0; i < 3; ++i) {
                int idx = tid + i * 256;         // 0..767 x 16B
                int n = idx >> 2;
                int c = idx & 3;
                async_b128_to_lds(wtg + (size_t)n * K + k0 + c * 8,
                                  &Bs[n * BSTR + c * 8]);
            }
            if (k0 + 32 < K)   // pull next weight block toward the caches
                __builtin_prefetch(wtg + (size_t)(tid >> 2) * K + k0 + 32 +
                                       (tid & 3) * 8, 0, 1);
        }
        __syncthreads();
        // ---- stage A (im2col 64x32): thread -> row tid>>2, 8 k-values ----
        {
            int row = tid >> 2;
            int kk0 = (tid & 3) * 8;
            int oy = row / tw;
            int ox = row - oy * tw;
            Pack8 tv;
            #pragma unroll
            for (int j = 0; j < 8; ++j) {
                int kk = kk0 + j;
                f16 val = (f16)0.f;
                if (row < MT) {
                    int iy = oy + s_dy[kk] - 1;
                    int ix = ox + s_dx[kk] - 1;
                    if (iy >= 0 && iy < th && ix >= 0 && ix < tw)
                        val = hin[hbase + ((size_t)s_ic[kk] * H + (ty0 + iy)) * W +
                                  (tx0 + ix)];
                }
                tv.h[j] = val;
            }
            *(uint4*)&As[row * ASTR + kk0] = tv.q;
        }
#if HAVE_ASYNC_LDS
        wait_async_done();                   // this wave's B DMA landed
#endif
        __syncthreads();
        // ---- fragments + 6 WMMAs (all 16B LDS loads) ----
        Frag16 ua;
        {
            int row = mt * 16 + lm;
            int r0 = hi ? 8 : 0;
            ua.q[0] = *(const uint4*)&As[row * ASTR + r0];
            ua.q[1] = *(const uint4*)&As[row * ASTR + 16 + r0];
        }
        int kb = hi ? 8 : 0;
        #pragma unroll
        for (int t = 0; t < 6; ++t) {
            int n = (ntb + t) * 16 + lm;
            Frag16 ub;
            ub.q[0] = *(const uint4*)&Bs[n * BSTR + kb];
            ub.q[1] = *(const uint4*)&Bs[n * BSTR + 16 + kb];
            acc[t] = __builtin_amdgcn_wmma_f32_16x16x32_f16(
                false, ua.v, false, ub.v, (short)0, acc[t], false, false);
        }
    }

    // ---- epilogue: bias + GELU + residual, store f16 ----
    int mrow = mt * 16 + (hi ? 8 : 0);
    #pragma unroll
    for (int t = 0; t < 6; ++t) {
        int n = (ntb + t) * 16 + lm;
        float bb = bias[(size_t)g * C + n];
        #pragma unroll
        for (int r = 0; r < 8; ++r) {
            int m = mrow + r;
            if (m < MT) {
                int oy = m / tw;
                int ox = m - oy * tw;
                size_t oidx = hbase + ((size_t)n * H + (ty0 + oy)) * W + (tx0 + ox);
                float y = gelu_exact(acc[t][r] + bb);
                y += (float)hin[oidx];
                hout[oidx] = (f16)y;
            }
        }
    }
}

// ---------------------------------------------------------------------------
// Downsampling 2x2/s2 conv as implicit GEMM (K = Cin*4).
// blockIdx = (m-block of 64 rows, n-block of 192 cols, batch).
// DO_GELU: apply BN+GELU, store f16 NCHW.  TR_F32: store f32 (b, m, n).
// ---------------------------------------------------------------------------
template <bool DO_GELU, bool TR_F32>
__global__ __launch_bounds__(256)
void down_conv(const f16* __restrict__ hin, const f16* __restrict__ wt,
               const float* __restrict__ bias, f16* __restrict__ hout16,
               float* __restrict__ houtf, int Cin, int Hin, int Win, int OC) {
    const int HO = Hin >> 1, WO = Win >> 1;
    const int M = HO * WO;
    const int K = Cin * 4;
    const int mb = blockIdx.x, nb = blockIdx.y, b = blockIdx.z;
    const int tid  = threadIdx.x;
    const int lane = tid & 31;
    const int wid  = tid >> 5;
    const int lm   = lane & 15;
    const bool hi  = (lane & 16) != 0;

    __shared__ __align__(16) f16 As[64 * ASTR];
    __shared__ __align__(16) f16 Bs[192 * BSTR];

    const int mt  = wid >> 1;
    const int ntb = (wid & 1) * 6;

    v8f acc[6];
    #pragma unroll
    for (int t = 0; t < 6; ++t) {
        v8f z = {0.f, 0.f, 0.f, 0.f, 0.f, 0.f, 0.f, 0.f};
        acc[t] = z;
    }

    const size_t hinb = (size_t)b * Cin * Hin * Win;

    for (int k0 = 0; k0 < K; k0 += 32) {
        __syncthreads();
        // ---- stage B: async DMA 192 rows x 32 k-contiguous to LDS ----
        {
            #pragma unroll
            for (int i = 0; i < 3; ++i) {
                int idx = tid + i * 256;        // 0..767 x 16B
                int n = idx >> 2;
                int c = idx & 3;
                async_b128_to_lds(wt + (size_t)(nb * 192 + n) * K + k0 + c * 8,
                                  &Bs[n * BSTR + c * 8]);
            }
            if (k0 + 32 < K)
                __builtin_prefetch(wt + (size_t)(nb * 192 + (tid >> 2)) * K +
                                       k0 + 32 + (tid & 3) * 8, 0, 1);
        }
        // ---- stage A ----
        {
            int row = tid >> 2;
            int kk0 = (tid & 3) * 8;
            int m = mb * 64 + row;
            int oy = m / WO;
            int ox = m - oy * WO;
            Pack8 tv;
            #pragma unroll
            for (int j = 0; j < 8; ++j) {
                int k = k0 + kk0 + j;
                f16 val = (f16)0.f;
                if (m < M) {
                    int ic = k >> 2;
                    int dy = (k >> 1) & 1;
                    int dx = k & 1;
                    val = hin[hinb + ((size_t)ic * Hin + (2 * oy + dy)) * Win +
                              (2 * ox + dx)];
                }
                tv.h[j] = val;
            }
            *(uint4*)&As[row * ASTR + kk0] = tv.q;
        }
#if HAVE_ASYNC_LDS
        wait_async_done();                  // this wave's B DMA landed
#endif
        __syncthreads();
        // ---- fragments + WMMA ----
        Frag16 ua;
        {
            int row = mt * 16 + lm;
            int r0 = hi ? 8 : 0;
            ua.q[0] = *(const uint4*)&As[row * ASTR + r0];
            ua.q[1] = *(const uint4*)&As[row * ASTR + 16 + r0];
        }
        int kb = hi ? 8 : 0;
        #pragma unroll
        for (int t = 0; t < 6; ++t) {
            int n = (ntb + t) * 16 + lm;
            Frag16 ub;
            ub.q[0] = *(const uint4*)&Bs[n * BSTR + kb];
            ub.q[1] = *(const uint4*)&Bs[n * BSTR + 16 + kb];
            acc[t] = __builtin_amdgcn_wmma_f32_16x16x32_f16(
                false, ua.v, false, ub.v, (short)0, acc[t], false, false);
        }
    }

    // ---- epilogue ----
    int mrow = mb * 64 + mt * 16 + (hi ? 8 : 0);
    #pragma unroll
    for (int t = 0; t < 6; ++t) {
        int n = nb * 192 + (ntb + t) * 16 + lm;
        float bb = bias[n];
        #pragma unroll
        for (int r = 0; r < 8; ++r) {
            int m = mrow + r;
            if (m < M) {
                float y = acc[t][r] + bb;
                if (DO_GELU) y = gelu_exact(y);
                if (TR_F32) {
                    houtf[((size_t)b * M + m) * OC + n] = y;
                } else {
                    int oy = m / WO;
                    int ox = m - oy * WO;
                    hout16[((size_t)(b * OC + n) * HO + oy) * WO + ox] = (f16)y;
                }
            }
        }
    }
}

// ---------------------------------------------------------------------------
extern "C" void kernel_launch(void* const* d_in, const int* in_sizes, int n_in,
                              void* d_out, int out_size, void* d_ws, size_t ws_size,
                              hipStream_t stream) {
    (void)in_sizes; (void)n_in; (void)out_size; (void)ws_size;

    const float* x       = (const float*)d_in[0];
    const float* c1w     = (const float*)d_in[1];
    const float* bn1[4]  = {(const float*)d_in[2], (const float*)d_in[3],
                            (const float*)d_in[4], (const float*)d_in[5]};
    const float* r1w     = (const float*)d_in[6];
    const float* rb1[4]  = {(const float*)d_in[7], (const float*)d_in[8],
                            (const float*)d_in[9], (const float*)d_in[10]};
    const float* c2w     = (const float*)d_in[11];
    const float* bn2[4]  = {(const float*)d_in[12], (const float*)d_in[13],
                            (const float*)d_in[14], (const float*)d_in[15]};
    const float* r2w     = (const float*)d_in[16];
    const float* rb2[4]  = {(const float*)d_in[17], (const float*)d_in[18],
                            (const float*)d_in[19], (const float*)d_in[20]};
    const float* c3w     = (const float*)d_in[21];
    const float* bn3[4]  = {(const float*)d_in[22], (const float*)d_in[23],
                            (const float*)d_in[24], (const float*)d_in[25]};

    const size_t A1 = (size_t)64 * 192 * 56 * 56;   // stage-1 activations
    const size_t A2 = (size_t)64 * 192 * 28 * 28;   // stage-2 activations

    size_t off = 0;
    auto alloc = [&](size_t bytes) {
        off = (off + 255) & ~(size_t)255;
        size_t o = off; off += bytes; return o;
    };
    char* ws = (char*)d_ws;
    f16*   h1a = (f16*)(ws + alloc(A1 * 2));
    f16*   h1b = (f16*)(ws + alloc(A1 * 2));
    f16*   h2a = (f16*)(ws + alloc(A2 * 2));
    f16*   h2b = (f16*)(ws + alloc(A2 * 2));
    f16*   w1t = (f16*)(ws + alloc((size_t)48 * 192 * 2));
    f16*   wr1 = (f16*)(ws + alloc((size_t)64 * 1728 * 192 * 2));
    f16*   w2t = (f16*)(ws + alloc((size_t)768 * 192 * 2));
    f16*   wr2 = (f16*)(ws + alloc((size_t)16 * 1728 * 192 * 2));
    f16*   w3t = (f16*)(ws + alloc((size_t)768 * 768 * 2));
    float* b1  = (float*)(ws + alloc(192 * 4));
    float* br1 = (float*)(ws + alloc((size_t)64 * 192 * 4));
    float* b2  = (float*)(ws + alloc(192 * 4));
    float* br2 = (float*)(ws + alloc((size_t)16 * 192 * 4));
    float* b3  = (float*)(ws + alloc(768 * 4));

    auto blocks = [](size_t total) { return (unsigned)((total + 255) / 256); };

    // Fold BN into weights (f16, original (G,OC,K) layout) + biases.
    fold_weights<<<blocks((size_t)1 * 192 * 48), 256, 0, stream>>>(
        c1w, bn1[0], bn1[1], bn1[2], bn1[3], w1t, b1, 1, 192, 48);
    fold_weights<<<blocks((size_t)64 * 192 * 1728), 256, 0, stream>>>(
        r1w, rb1[0], rb1[1], rb1[2], rb1[3], wr1, br1, 64, 192, 1728);
    fold_weights<<<blocks((size_t)1 * 192 * 768), 256, 0, stream>>>(
        c2w, bn2[0], bn2[1], bn2[2], bn2[3], w2t, b2, 1, 192, 768);
    fold_weights<<<blocks((size_t)16 * 192 * 1728), 256, 0, stream>>>(
        r2w, rb2[0], rb2[1], rb2[2], rb2[3], wr2, br2, 16, 192, 1728);
    fold_weights<<<blocks((size_t)1 * 768 * 768), 256, 0, stream>>>(
        c3w, bn3[0], bn3[1], bn3[2], bn3[3], w3t, b3, 1, 768, 768);

    // Stage 1: 4x4/s4 conv + BN + GELU -> (64,192,56,56) f16.
    conv1_kernel<<<blocks(A1), 256, 0, stream>>>(x, w1t, b1, h1a);

    // Region 1: 8x8 grid of per-tile 3x3 convs (WMMA implicit GEMM).
    region_conv<<<dim3(64, 64), 256, 0, stream>>>(h1a, wr1, br1, h1b,
                                                  192, 56, 56, 8, 8);

    // Stage 2: 2x2/s2 conv + BN + GELU -> (64,192,28,28) f16.
    down_conv<true, false><<<dim3(13, 1, 64), 256, 0, stream>>>(
        h1b, w2t, b2, h2a, nullptr, 192, 56, 56, 192);

    // Region 2: 4x4 grid of per-tile 3x3 convs.
    region_conv<<<dim3(16, 64), 256, 0, stream>>>(h2a, wr2, br2, h2b,
                                                  192, 28, 28, 4, 4);

    // Stage 3: 2x2/s2 conv 192->768 + BN, store transposed f32 (64,196,768).
    down_conv<false, true><<<dim3(4, 4, 64), 256, 0, stream>>>(
        h2b, w3t, b3, nullptr, (float*)d_out, 192, 28, 28, 768);
}
